// GNNBase_72129680769714
// MI455X (gfx1250) — matrix operand order repaired and checked by reading, not compiled, not run on hardware
//
#include <hip/hip_runtime.h>
#include <hip/hip_bf16.h>

// ---------------------------------------------------------------------------
// GNN (GAT-style) forward for MI455X / gfx1250, wave32 + WMMA bf16.
//
// v4: fix TDM builtin arity — this toolchain exposes the 6-argument
// __builtin_amdgcn_tensor_load_to_lds(g0 v4u, g1 v8i, v4i, v4i, v8i, i32 cpol).
//
// B-slab staging (32 x K bf16 of transposed weights -> LDS) uses the CDNA5
// Tensor Data Mover: one wave issues TENSOR_LOAD_TO_LDS with a 2D D#
// descriptor, waits on TENSORcnt, then the block barriers.
//
// Hot loop: per wave, 2x global_load_b128 (bf16 A frag, gathered row),
// 4x ds_load_b128 (two B frags from LDS), 2x v_wmma_f32_16x16x32_bf16 ->
// two 16x16 C tiles per A fragment; B traffic amortized 8x across the
// block's waves. Edge GEMM collapsed to a 5xOC LUT fused into the epilogue.
// ---------------------------------------------------------------------------

#define BATCH 4
#define NN    4096
#define EE    16384
#define EDIM  512

typedef __attribute__((ext_vector_type(16))) __bf16       v16bf;
typedef __attribute__((ext_vector_type(8)))  __bf16       v8bf;
typedef __attribute__((ext_vector_type(8)))  float        v8f;
typedef __attribute__((ext_vector_type(4)))  unsigned int v4u;
typedef __attribute__((ext_vector_type(8)))  int          v8i;
typedef __attribute__((ext_vector_type(4)))  int          v4i;

// ------------------------- small helpers -----------------------------------

__device__ __forceinline__ void atomic_max_float(int* addr, float v) {
    int iv  = __float_as_int(v);
    int key = (iv >= 0) ? iv : (iv ^ 0x7FFFFFFF);   // order-preserving encode
    atomicMax(addr, key);
}
__device__ __forceinline__ float decode_max(int key) {
    return (key >= 0) ? __int_as_float(key) : __int_as_float(key ^ 0x7FFFFFFF);
}

__global__ void k_fill_f(float* p, int n, float v) {
    int i = blockIdx.x * blockDim.x + threadIdx.x;
    if (i < n) p[i] = v;
}
__global__ void k_fill_i(int* p, int n, int v) {
    int i = blockIdx.x * blockDim.x + threadIdx.x;
    if (i < n) p[i] = v;
}

// Wt[n*K + k] = bf16(W[k*OC + n])  -- coalesced writes along k
__global__ void k_transpose_bf16(const float* __restrict__ W,
                                 __bf16* __restrict__ Wt, int K, int OC) {
    int i = blockIdx.x * blockDim.x + threadIdx.x;
    if (i >= K * OC) return;
    int n = i / K, k = i % K;
    Wt[i] = (__bf16)W[(size_t)k * OC + n];
}

// ------------------------- embedding ---------------------------------------

__global__ void k_embed(const int* __restrict__ nodes,
                        const int* __restrict__ assignment,
                        const float* __restrict__ node_emb,
                        const float* __restrict__ assign_emb,
                        float* __restrict__ h, __bf16* __restrict__ hb, int b) {
    int i = blockIdx.x * blockDim.x + threadIdx.x;   // N*512
    int n = i >> 9, d = i & 511;
    int nid = nodes[b * NN + n] + 1;
    float v = node_emb[nid * EDIM + d] + assign_emb[assignment[b] * EDIM + d];
    h[i]  = v;
    hb[i] = (__bf16)v;
}

// lut[t, j] = b_edge[j] + sum_d (edge_emb[t,d] + asg[d]) * W_edge[d, j]
__global__ void k_lut(const float* __restrict__ edge_emb,
                      const float* __restrict__ assign_emb,
                      const int* __restrict__ assignment,
                      const float* __restrict__ W_edge,
                      const float* __restrict__ b_edge,
                      float* __restrict__ lut, int OC, int b) {
    int i = blockIdx.x * blockDim.x + threadIdx.x;
    if (i >= 5 * OC) return;
    int t = i / OC, j = i % OC;
    const float* asg = assign_emb + assignment[b] * EDIM;
    float s = b_edge[j];
    for (int d = 0; d < EDIM; ++d)
        s += (edge_emb[t * EDIM + d] + asg[d]) * W_edge[d * OC + j];
    lut[i] = s;
}

// ------------------------- WMMA GEMM ---------------------------------------
// out[m, n0..n0+31] = A[row(m), :K] @ Wt[n0..n0+31, :K]^T + bias + lut[etype]
//   A: bf16 rows length K.  Wt: bf16, OC rows x K (transposed weights).
//   Block: 256 threads = 8 waves. Wave w owns M-tile (blockIdx.x*8 + w),
//   all waves share the 32-column slab Bs (TDM-staged in LDS), 2 N-tiles/wave.
__global__ void __launch_bounds__(256)
k_gemm_bf16_wmma(const __bf16* __restrict__ A, int K,
                 const int* __restrict__ gidx, int gstride,
                 const __bf16* __restrict__ Wt, int OC,
                 const float* __restrict__ bias,
                 const float* __restrict__ lut,
                 const int* __restrict__ etyp,
                 float* __restrict__ out) {
    __shared__ __align__(128) __bf16 Bs[32 * 512];   // 32 cols x K (K<=512), 32KB

    const int n0 = blockIdx.y * 32;

#if defined(__has_builtin) && __has_builtin(__builtin_amdgcn_tensor_load_to_lds)
    // --- Tensor Data Mover staging: one 2D tile DMA issued by wave 0 ---
    if (threadIdx.x < 32) {
        // D# group 0: [1:0]=count=1 | [63:32]=lds_addr | [120:64]=global_addr
        //             [127:126]=type=2
        unsigned long long ga = (unsigned long long)(const void*)(Wt + (size_t)n0 * K);
        unsigned int lds = (unsigned int)(unsigned long long)(const void*)&Bs[0];
        v4u g0;
        g0[0] = 1u;                                   // count=1, user desc
        g0[1] = lds;                                  // LDS byte address
        g0[2] = (unsigned int)(ga & 0xFFFFFFFFu);     // global addr [31:0]
        g0[3] = (unsigned int)((ga >> 32) & 0x01FFFFFFu) | (2u << 30); // [56:32]|type
        // D# group 1: data_size=1 (2B), tensor = OC rows x K elems (row stride K),
        //             tile = 32 rows x K elems
        v8i g1;
        g1[0] = (1 << 16);                            // data_size=2B; mask=0
        g1[1] = (int)((K & 0xFFFF) << 16);            // tensor_dim0[15:0]
        g1[2] = (int)((K >> 16) & 0xFFFF) | (int)((OC & 0xFFFF) << 16); // td0 hi|td1 lo
        g1[3] = (int)((OC >> 16) & 0xFFFF) | (int)((K & 0xFFFF) << 16); // td1 hi|tile_dim0
        g1[4] = 32;                                   // tile_dim1=32, tile_dim2=0
        g1[5] = K;                                    // tensor_dim0_stride[31:0]
        g1[6] = 0;                                    // stride hi | dim1_stride lo
        g1[7] = 0;
        v4i z4 = {0, 0, 0, 0};                        // groups 2/3: unused (2D)
        v8i z8 = {0, 0, 0, 0, 0, 0, 0, 0};
        __builtin_amdgcn_tensor_load_to_lds(g0, g1, z4, z4, z8, 0);
        __builtin_amdgcn_s_wait_tensorcnt((short)0);  // TENSORcnt -> 0
    }
#else
    // --- fallback: cooperative vectorized copy (4*K 16B chunks) ---
    const int nchunks = 4 * K;
    for (int t = threadIdx.x; t < nchunks; t += 256) {
        int base = t * 8;
        int r = base / K, k = base % K;
        *(v8bf*)&Bs[r * K + k] = *(const v8bf*)&Wt[(size_t)(n0 + r) * K + k];
    }
#endif
    __syncthreads();

    const int w    = threadIdx.x >> 5;               // wave in block
    const int lane = threadIdx.x & 31;
    const int hv   = lane >> 4;                      // half-wave
    const int lp   = lane & 15;

    const int tile_m = blockIdx.x * 8 + w;
    const int m      = tile_m * 16 + lp;             // A-fragment row
    const int arow   = gidx ? gidx[(size_t)m * gstride] : m;
    const __bf16* __restrict__ Arow = A + (size_t)arow * K;

    // LDS pointers for the two N-tiles (col = lp and 16+lp), K-major
    const __bf16* b0p = &Bs[(size_t)lp * K];
    const __bf16* b1p = &Bs[(size_t)(16 + lp) * K];

    v8f acc0 = {}, acc1 = {};
    for (int k0 = 0; k0 < K; k0 += 32) {
        // A 16x32 layout: lane half hv holds K = 8*hv+0..7 and 16+8*hv+0..7
        v8bf alo = *(const v8bf*)&Arow[k0 + 8 * hv];
        v8bf ahi = *(const v8bf*)&Arow[k0 + 16 + 8 * hv];
        v16bf af = __builtin_shufflevector(alo, ahi,
            0, 1, 2, 3, 4, 5, 6, 7, 8, 9, 10, 11, 12, 13, 14, 15);
        // B 32x16 layout: lane half hv holds K = 8*hv+0..7 and 16+8*hv+0..7 for its column
        v8bf b0l = *(const v8bf*)&b0p[k0 + 8 * hv];
        v8bf b0h = *(const v8bf*)&b0p[k0 + 16 + 8 * hv];
        v8bf b1l = *(const v8bf*)&b1p[k0 + 8 * hv];
        v8bf b1h = *(const v8bf*)&b1p[k0 + 16 + 8 * hv];
        v16bf bf0 = __builtin_shufflevector(b0l, b0h,
            0, 1, 2, 3, 4, 5, 6, 7, 8, 9, 10, 11, 12, 13, 14, 15);
        v16bf bf1 = __builtin_shufflevector(b1l, b1h,
            0, 1, 2, 3, 4, 5, 6, 7, 8, 9, 10, 11, 12, 13, 14, 15);
        if (k0 + 32 < K)
            __builtin_prefetch(&Arow[k0 + 32], 0, 1);          // global_prefetch_b8
        acc0 = __builtin_amdgcn_wmma_f32_16x16x32_bf16(
            false, af, false, bf0, (short)0, acc0, false, false);
        acc1 = __builtin_amdgcn_wmma_f32_16x16x32_bf16(
            false, af, false, bf1, (short)0, acc1, false, false);
    }

    // C/D layout: VGPR r: lanes 0-15 -> M=r, lanes 16-31 -> M=8+r; N = lane%16
    const float bc0 = bias ? bias[n0 + lp]      : 0.0f;
    const float bc1 = bias ? bias[n0 + 16 + lp] : 0.0f;
#pragma unroll
    for (int r = 0; r < 8; ++r) {
        int mo = tile_m * 16 + r + 8 * hv;
        float a0 = bc0, a1 = bc1;
        if (lut) {
            int t = etyp[(size_t)mo * 3 + 2] + 1;
            const float* lrow = lut + t * OC;
            a0 += lrow[n0 + lp];
            a1 += lrow[n0 + 16 + lp];
        }
        out[(size_t)mo * OC + n0 + lp]      = acc0[r] + a0;
        out[(size_t)mo * OC + n0 + 16 + lp] = acc1[r] + a1;
    }
}

// ------------------------- attention / softmax -----------------------------

__global__ void k_alpha(const float* __restrict__ msg,
                        const float* __restrict__ att,
                        const int* __restrict__ edges_b,
                        float* __restrict__ exb, int* __restrict__ amax,
                        int hd, int oc) {
    int i = blockIdx.x * blockDim.x + threadIdx.x;
    if (i >= EE * hd) return;
    int e = i / hd, hh = i % hd;
    int OC = hd * oc;
    const float* mrow = msg + (size_t)e * OC + hh * oc;
    const float* arow = att + hh * oc;
    float s = 0.0f;
    for (int o = 0; o < oc; ++o) s += mrow[o] * arow[o];
    s = (s > 0.0f) ? s : 0.2f * s;                   // leaky_relu 0.2
    exb[i] = s;
    int dst = edges_b[e * 3 + 1];
    atomic_max_float(&amax[dst * hd + hh], s);
}

__global__ void k_expden(float* __restrict__ exb,
                         const int* __restrict__ amax,
                         const int* __restrict__ edges_b,
                         float* __restrict__ den, int hd) {
    int i = blockIdx.x * blockDim.x + threadIdx.x;
    if (i >= EE * hd) return;
    int e = i / hd, hh = i % hd;
    int dst = edges_b[e * 3 + 1];
    float ex = expf(exb[i] - decode_max(amax[dst * hd + hh]));
    exb[i] = ex;
    atomicAdd(&den[dst * hd + hh], ex);
}

__global__ void k_scatter(const float* __restrict__ msg,
                          const float* __restrict__ exb,
                          const float* __restrict__ den,
                          const int* __restrict__ edges_b,
                          float* __restrict__ acc, int hd, int oc) {
    int i = blockIdx.x * blockDim.x + threadIdx.x;
    if (i >= EE * hd) return;
    int e = i / hd, hh = i % hd;
    int dst = edges_b[e * 3 + 1];
    float wgt = exb[i] / (den[dst * hd + hh] + 1e-16f);
    int OC = hd * oc;
    const float* mrow = msg + (size_t)e * OC + hh * oc;
    float* arow = acc + (size_t)dst * OC + hh * oc;
    for (int o = 0; o < oc; ++o) atomicAdd(&arow[o], mrow[o] * wgt);
}

// hout[n,o] = (mean_h acc[n,h,o]) + resid[n,o]; optional ELU; bf16 mirror
__global__ void k_combine(const float* __restrict__ acc,
                          const float* __restrict__ resid, int rstride,
                          float* __restrict__ hout, __bf16* __restrict__ houtb,
                          int hd, int oc, int apply_elu) {
    int i = blockIdx.x * blockDim.x + threadIdx.x;
    if (i >= NN * oc) return;
    int n = i / oc, o = i % oc;
    int OC = hd * oc;
    float s = 0.0f;
    for (int hh = 0; hh < hd; ++hh) s += acc[(size_t)n * OC + hh * oc + o];
    s = s / (float)hd + resid[(size_t)n * rstride + o];
    if (apply_elu) s = (s > 0.0f) ? s : (expf(s) - 1.0f);
    hout[i]  = s;
    houtb[i] = (__bf16)s;
}

// feat = h[cursor]; critic = feat @ critic_W + critic_b
// d_out layout: [critic(B)] then [feat(B,32)]
__global__ void k_final(const float* __restrict__ h,
                        const int* __restrict__ cursor,
                        const float* __restrict__ cW,
                        const float* __restrict__ cb,
                        float* __restrict__ out, int b) {
    int j = threadIdx.x;                             // 32 lanes
    int cur = cursor[b];
    float f = h[cur * 32 + j];
    out[BATCH + b * 32 + j] = f;
    float p = f * cW[j];
    for (int offs = 16; offs > 0; offs >>= 1) p += __shfl_down(p, offs);
    if (j == 0) out[b] = p + cb[0];
}

// ------------------------- host driver -------------------------------------

extern "C" void kernel_launch(void* const* d_in, const int* in_sizes, int n_in,
                              void* d_out, int out_size, void* d_ws, size_t ws_size,
                              hipStream_t stream) {
    (void)in_sizes; (void)n_in; (void)out_size; (void)ws_size;

    const int*   nodes      = (const int*)d_in[0];   // (B,N)
    const int*   edges      = (const int*)d_in[1];   // (B,E,3) src,dst,eat
    const int*   assignment = (const int*)d_in[2];   // (B,)
    const int*   cursor     = (const int*)d_in[3];   // (B,)
    const float* node_emb   = (const float*)d_in[4]; // (50,512)
    const float* edge_emb   = (const float*)d_in[5]; // (5,512)
    const float* assign_emb = (const float*)d_in[6]; // (1,512)

    struct LCfg { int ic, oc, hd, ue, self_; };
    const LCfg L[4] = {{512,128,8,1,1},{128,64,8,1,1},{64,64,16,1,0},{64,32,1,0,1}};

    const float *Wm[4], *bm[4], *We[4], *be[4], *att[4], *Ws[4], *bs[4];
    int idx = 7;
    for (int l = 0; l < 4; ++l) {
        Wm[l] = (const float*)d_in[idx++];
        bm[l] = (const float*)d_in[idx++];
        if (L[l].ue) { We[l] = (const float*)d_in[idx++]; be[l] = (const float*)d_in[idx++]; }
        else         { We[l] = nullptr;                   be[l] = nullptr; }
        att[l] = (const float*)d_in[idx++];
        if (L[l].self_) { Ws[l] = (const float*)d_in[idx++]; bs[l] = (const float*)d_in[idx++]; }
        else            { Ws[l] = nullptr;                   bs[l] = nullptr; }
    }
    const float* cW = (const float*)d_in[idx++];
    const float* cb = (const float*)d_in[idx++];

    // ---- workspace carve (bytes, 256-aligned blocks) ----
    char* base = (char*)d_ws;
    size_t off = 0;
    auto carve = [&](size_t bytes) {
        void* p = base + off;
        off += (bytes + 255) & ~(size_t)255;
        return p;
    };
    float*  hA    = (float*) carve((size_t)NN * 512 * 4);
    float*  hB    = (float*) carve((size_t)NN * 512 * 4);
    __bf16* hAb   = (__bf16*)carve((size_t)NN * 512 * 2);
    __bf16* hBb   = (__bf16*)carve((size_t)NN * 512 * 2);
    float*  msg   = (float*) carve((size_t)EE * 1024 * 4);
    float*  exb   = (float*) carve((size_t)EE * 16 * 4);
    int*    amax  = (int*)   carve((size_t)NN * 16 * 4);
    float*  den   = (float*) carve((size_t)NN * 16 * 4);
    float*  acc   = (float*) carve((size_t)NN * 1024 * 4);
    float*  selfb = (float*) carve((size_t)NN * 128 * 4);
    float*  lut   = (float*) carve(5 * 1024 * 4);
    __bf16* WmT[4]; __bf16* WsT[4];
    for (int l = 0; l < 4; ++l) {
        int K = L[l].ic, OC = L[l].oc * L[l].hd;
        WmT[l] = (__bf16*)carve((size_t)K * OC * 2);
        WsT[l] = L[l].self_ ? (__bf16*)carve((size_t)K * L[l].oc * 2) : nullptr;
    }

    // ---- one-time (per call): transpose+convert weights to bf16 ----
    for (int l = 0; l < 4; ++l) {
        int K = L[l].ic, OC = L[l].oc * L[l].hd;
        k_transpose_bf16<<<(K * OC + 255) / 256, 256, 0, stream>>>(Wm[l], WmT[l], K, OC);
        if (L[l].self_)
            k_transpose_bf16<<<(K * L[l].oc + 255) / 256, 256, 0, stream>>>(Ws[l], WsT[l], K, L[l].oc);
    }

    for (int b = 0; b < BATCH; ++b) {
        const int* eb = edges + (size_t)b * EE * 3;

        k_embed<<<NN * 512 / 256, 256, 0, stream>>>(nodes, assignment, node_emb,
                                                    assign_emb, hA, hAb, b);
        float*  hin  = hA;  float*  hout  = hB;
        __bf16* hinb = hAb; __bf16* houtb = hBb;

        for (int l = 0; l < 4; ++l) {
            const int ic = L[l].ic, oc = L[l].oc, hd = L[l].hd;
            const int OC = oc * hd;

            if (L[l].ue)
                k_lut<<<(5 * OC + 255) / 256, 256, 0, stream>>>(
                    edge_emb, assign_emb, assignment, We[l], be[l], lut, OC, b);

            k_fill_i<<<(NN * hd + 255) / 256, 256, 0, stream>>>(amax, NN * hd, (int)0x80000000);
            k_fill_f<<<(NN * hd + 255) / 256, 256, 0, stream>>>(den, NN * hd, 0.0f);
            k_fill_f<<<(NN * OC + 255) / 256, 256, 0, stream>>>(acc, NN * OC, 0.0f);

            // msg = h[src] @ W_msg + b_msg + lut[etype]   (WMMA bf16, TDM-staged B)
            k_gemm_bf16_wmma<<<dim3(EE / 128, OC / 32), 256, 0, stream>>>(
                hinb, ic, eb, 3, WmT[l], OC, bm[l],
                L[l].ue ? lut : nullptr, eb, msg);

            k_alpha<<<(EE * hd + 255) / 256, 256, 0, stream>>>(msg, att[l], eb, exb, amax, hd, oc);
            k_expden<<<(EE * hd + 255) / 256, 256, 0, stream>>>(exb, amax, eb, den, hd);
            k_scatter<<<(EE * hd + 255) / 256, 256, 0, stream>>>(msg, exb, den, eb, acc, hd, oc);

            const float* resid; int rstride;
            if (L[l].self_) {
                k_gemm_bf16_wmma<<<dim3(NN / 128, oc / 32), 256, 0, stream>>>(
                    hinb, ic, nullptr, 0, WsT[l], oc, bs[l], nullptr, nullptr, selfb);
                resid = selfb; rstride = oc;
            } else {
                resid = hin; rstride = ic;
            }

            k_combine<<<(NN * oc + 255) / 256, 256, 0, stream>>>(
                acc, resid, rstride, hout, houtb, hd, oc, (l < 3) ? 1 : 0);

            float*  tf = hin;  hin  = hout;  hout  = tf;
            __bf16* tb = hinb; hinb = houtb; houtb = tb;
        }
        k_final<<<1, 32, 0, stream>>>(hin, cursor, cW, cb, (float*)d_out, b);
    }
}
